// MultiHeadAttentionBlock_44255343018898
// MI455X (gfx1250) — matrix operand order repaired and checked by reading, not compiled, not run on hardware
//
#include <hip/hip_runtime.h>
#include <hip/hip_bf16.h>

// Problem constants (from reference)
#define B_  2
#define S_  2048
#define DM  768
#define NH  8
#define DK  96
// 1/sqrt(96)
#define RSCALE 0.1020620726159658f

typedef __bf16 bf16_t;
typedef __attribute__((ext_vector_type(16))) __bf16 v16bf;
typedef __attribute__((ext_vector_type(8)))  __bf16 v8bf;
typedef __attribute__((ext_vector_type(8)))  float  v8f;

// ---------------- WMMA fragment loaders (wave32, 16x16x32 bf16) ----------------
// A (16x32, MxK): lane row = lane&15; elem j -> K = 16*(j>>3) + 8*(lane>>4) + (j&7)
__device__ __forceinline__ v16bf frag_a_bf(const bf16_t* prow, int k0, int lane) {
  const int base = (lane & 16) ? 8 : 0;
  const bf16_t* p = prow + k0 + base;
  v8bf lo = *(const v8bf*)(p);
  v8bf hi = *(const v8bf*)(p + 16);
  v16bf f;
#pragma unroll
  for (int i = 0; i < 8; ++i) { f[i] = lo[i]; f[8 + i] = hi[i]; }
  return f;
}

// B (32x16, KxN), sourced from Bt[n][k] (K contiguous): lane col = lane&15;
// elem j -> K = 16*(j>>3) + 8*(lane>>4) + (j&7)
__device__ __forceinline__ v16bf frag_b_bf(const bf16_t* prow, int k0, int lane) {
  const int kb = (lane & 16) ? 8 : 0;
  const bf16_t* p = prow + k0 + kb;
  v8bf lo = *(const v8bf*)(p);
  v8bf hi = *(const v8bf*)(p + 16);
  v16bf f;
#pragma unroll
  for (int i = 0; i < 8; ++i) { f[i] = lo[i]; f[8 + i] = hi[i]; }
  return f;
}

// A fragment from fp32 source (attention probs), converted to bf16 in-register
__device__ __forceinline__ v16bf frag_a_f32(const float* prow, int k0, int lane) {
  const int base = (lane & 16) ? 8 : 0;
  const float* p = prow + k0 + base;
  v8f lo = *(const v8f*)(p);
  v8f hi = *(const v8f*)(p + 16);
  v16bf f;
#pragma unroll
  for (int i = 0; i < 8; ++i) { f[i] = (bf16_t)lo[i]; f[8 + i] = (bf16_t)hi[i]; }
  return f;
}

// ---------------- Conversions ----------------
__global__ __launch_bounds__(256) void cvt_f32_to_bf16(const float* __restrict__ src,
                                                       bf16_t* __restrict__ dst, int n) {
  int i = blockIdx.x * blockDim.x + threadIdx.x;
  if (i < n) dst[i] = (bf16_t)src[i];
}

// WT[w][n][k] = W_w[k][n]  (so B-fragment loads are K-contiguous)
__global__ __launch_bounds__(256) void cvt_weight_T(const float* __restrict__ Wq,
                                                    const float* __restrict__ Wk,
                                                    const float* __restrict__ Wv,
                                                    const float* __restrict__ Wo,
                                                    bf16_t* __restrict__ WT) {
  int i = blockIdx.x * blockDim.x + threadIdx.x;
  const int WSZ = DM * DM;
  if (i >= 4 * WSZ) return;
  int w = i / WSZ, r = i % WSZ;
  int n = r / DM, k = r % DM;
  const float* W = (w == 0) ? Wq : (w == 1) ? Wk : (w == 2) ? Wv : Wo;
  WT[i] = (bf16_t)W[(size_t)k * DM + n];
}

// Vt[b][h][d][s] = Vb[b*S+s][h*DK+d]
__global__ __launch_bounds__(256) void transpose_v_bf16(const bf16_t* __restrict__ Vb,
                                                        bf16_t* __restrict__ Vt) {
  int i = blockIdx.x * blockDim.x + threadIdx.x;
  if (i >= B_ * NH * DK * S_) return;
  int s = i % S_; int t = i / S_;
  int d = t % DK; t /= DK;
  int h = t % NH; int b = t / NH;
  Vt[i] = Vb[((size_t)(b * S_ + s)) * DM + h * DK + d];
}

// ---------------- GEMM: C[M,N] = A[M,K](bf16) x Bt[N,K](bf16) ----------------
// One wave computes a 16x64 strip (4 N-tiles). All fragments for a k-step are
// preloaded into distinct registers (one load clause, one wait), then the 4
// WMMAs issue back-to-back.
#define GEMM_NT 4
__global__ __launch_bounds__(256) void gemm_bf16_kernel(const bf16_t* __restrict__ A,
                                                        const bf16_t* __restrict__ Bt,
                                                        bf16_t* __restrict__ Cbf,
                                                        float* __restrict__ Cf,
                                                        int M, int N, int K, int store_f32) {
  const int lane = threadIdx.x & 31;
  const int gid  = blockIdx.x * (blockDim.x >> 5) + (threadIdx.x >> 5);
  const int grp_n  = N / (16 * GEMM_NT);
  const int ntiles = (M >> 4) * grp_n;
  if (gid >= ntiles) return;               // uniform per wave
  const int tm = gid / grp_n, tg = gid % grp_n;
  const bf16_t* prowA = A  + (size_t)(tm * 16 + (lane & 15)) * K;
  const bf16_t* prowB = Bt + (size_t)(tg * (16 * GEMM_NT) + (lane & 15)) * K;
  v8f acc[GEMM_NT] = {};
  for (int kk = 0; kk < K; kk += 32) {
    v16bf a = frag_a_bf(prowA, kk, lane);
    v16bf bfr[GEMM_NT];
#pragma unroll
    for (int j = 0; j < GEMM_NT; ++j)
      bfr[j] = frag_b_bf(prowB + (size_t)j * 16 * K, kk, lane);
#pragma unroll
    for (int j = 0; j < GEMM_NT; ++j)
      acc[j] = __builtin_amdgcn_wmma_f32_16x16x32_bf16(false, a, false, bfr[j], (short)0, acc[j], false, false);
  }
  const int hf = lane >> 4;
#pragma unroll
  for (int j = 0; j < GEMM_NT; ++j) {
    const int n = tg * (16 * GEMM_NT) + j * 16 + (lane & 15);
#pragma unroll
    for (int r = 0; r < 8; ++r) {
      const int m = tm * 16 + r + 8 * hf;  // C/D layout: row = r + 8*(lane>>4)
      if (store_f32) Cf[(size_t)m * N + n] = acc[j][r];
      else           Cbf[(size_t)m * N + n] = (bf16_t)acc[j][r];
    }
  }
}

// ---------------- Logits: attn[b,h,q,k] = mask ? (Q.Kt)/sqrt(dk) : -1e9 ----------------
// Store-bound (268 MB mandatory write) -> one 16x16 tile per wave is fine.
__global__ __launch_bounds__(256) void logits_kernel(const bf16_t* __restrict__ Qb,
                                                     const bf16_t* __restrict__ Kb,
                                                     const int* __restrict__ mask,
                                                     float* __restrict__ attn) {
  const int lane = threadIdx.x & 31;
  const int gid  = blockIdx.x * (blockDim.x >> 5) + (threadIdx.x >> 5);
  const int TQ = S_ / 16, TK = S_ / 16;
  int t = gid;
  const int kt = t % TK; t /= TK;
  const int qt = t % TQ; t /= TQ;
  const int h  = t % NH; t /= NH;
  const int b  = t;
  const bf16_t* prowA = Qb + (size_t)(b * S_ + qt * 16 + (lane & 15)) * DM;
  const bf16_t* prowB = Kb + (size_t)(b * S_ + kt * 16 + (lane & 15)) * DM;  // K row-major == Bt
  const int k0 = h * DK;
  v8f acc = {};
#pragma unroll
  for (int kk = 0; kk < DK; kk += 32) {   // 96 = 3 x 32
    v16bf a  = frag_a_bf(prowA, k0 + kk, lane);
    v16bf bb = frag_b_bf(prowB, k0 + kk, lane);
    acc = __builtin_amdgcn_wmma_f32_16x16x32_bf16(false, a, false, bb, (short)0, acc, false, false);
  }
  const int hf = lane >> 4, kc = kt * 16 + (lane & 15);
#pragma unroll
  for (int r = 0; r < 8; ++r) {
    const int q = qt * 16 + r + 8 * hf;
    float v = acc[r] * RSCALE;
    int m = mask[((size_t)b * S_ + q) * S_ + kc];
    attn[(((size_t)(b * NH + h)) * S_ + q) * S_ + kc] = (m == 0) ? -1e9f : v;
  }
}

// ---------------- Row softmax, in place, one block per row of 2048 ----------------
__global__ __launch_bounds__(256) void softmax_kernel(float* __restrict__ attn) {
  __shared__ float sm[256];
  const int row = blockIdx.x;
  float* p = attn + (size_t)row * S_;
  const int tid = threadIdx.x;
  float mx = -3.4e38f;
  for (int i = tid; i < S_; i += 256) mx = fmaxf(mx, p[i]);
  sm[tid] = mx; __syncthreads();
  for (int s = 128; s > 0; s >>= 1) { if (tid < s) sm[tid] = fmaxf(sm[tid], sm[tid + s]); __syncthreads(); }
  mx = sm[0]; __syncthreads();
  float sum = 0.0f;
  for (int i = tid; i < S_; i += 256) { float e = __expf(p[i] - mx); p[i] = e; sum += e; }
  sm[tid] = sum; __syncthreads();
  for (int s = 128; s > 0; s >>= 1) { if (tid < s) sm[tid] += sm[tid + s]; __syncthreads(); }
  const float inv = 1.0f / sm[0];
  for (int i = tid; i < S_; i += 256) p[i] *= inv;
}

// ---------------- AV: Ctx[b*S+q][h*DK+d] = sum_k attn[b,h,q,k] * V[b,k,h*DK+d] ----------------
// One wave computes the full 16x96 strip (all 6 d-tiles of a head): the fp32
// attn fragment (dominant HBM stream) is loaded+converted ONCE per k-step and
// reused 6x; all 6 B fragments preloaded so WMMAs issue back-to-back.
__global__ __launch_bounds__(256) void av_kernel(const float* __restrict__ attn,
                                                 const bf16_t* __restrict__ Vt,
                                                 bf16_t* __restrict__ Ctx) {
  const int lane = threadIdx.x & 31;
  const int gid  = blockIdx.x * (blockDim.x >> 5) + (threadIdx.x >> 5);
  const int TQ = S_ / 16, TD = DK / 16;   // 128, 6
  int t = gid;
  const int qt = t % TQ; t /= TQ;
  const int h  = t % NH; t /= NH;
  const int b  = t;
  const float*  prowA  = attn + ((size_t)(b * NH + h) * S_ + qt * 16 + (lane & 15)) * (size_t)S_;
  const bf16_t* prowB0 = Vt   + ((size_t)(b * NH + h) * DK + (lane & 15)) * (size_t)S_;
  v8f acc[TD] = {};
  for (int kk = 0; kk < S_; kk += 32) {   // 64 k-steps, 6 WMMAs each
    v16bf a = frag_a_f32(prowA, kk, lane);
    v16bf bfr[TD];
#pragma unroll
    for (int dt = 0; dt < TD; ++dt)
      bfr[dt] = frag_b_bf(prowB0 + (size_t)dt * 16 * S_, kk, lane);
#pragma unroll
    for (int dt = 0; dt < TD; ++dt)
      acc[dt] = __builtin_amdgcn_wmma_f32_16x16x32_bf16(false, a, false, bfr[dt], (short)0, acc[dt], false, false);
  }
  const int hf = lane >> 4;
#pragma unroll
  for (int dt = 0; dt < TD; ++dt) {
    const int nc = h * DK + dt * 16 + (lane & 15);
#pragma unroll
    for (int r = 0; r < 8; ++r) {
      const int m = b * S_ + qt * 16 + r + 8 * hf;
      Ctx[(size_t)m * DM + nc] = (bf16_t)acc[dt][r];
    }
  }
}

// ---------------- Host side ----------------
extern "C" void kernel_launch(void* const* d_in, const int* in_sizes, int n_in,
                              void* d_out, int out_size, void* d_ws, size_t ws_size,
                              hipStream_t stream) {
  (void)in_sizes; (void)n_in; (void)out_size; (void)ws_size;
  const float* q    = (const float*)d_in[0];
  const float* k    = (const float*)d_in[1];
  const float* v    = (const float*)d_in[2];
  const int*   mask = (const int*)  d_in[3];
  const float* Wq   = (const float*)d_in[4];
  const float* Wk   = (const float*)d_in[5];
  const float* Wv   = (const float*)d_in[6];
  const float* Wo   = (const float*)d_in[7];

  const size_t NX  = (size_t)B_ * S_ * DM;    // 3,145,728 elements
  const size_t SZX = NX * sizeof(bf16_t);     // 6.29 MB
  const size_t NW  = (size_t)DM * DM;
  const size_t SZW = NW * sizeof(bf16_t);

  char* ws = (char*)d_ws;
  bf16_t* Xq  = (bf16_t*)(ws);
  bf16_t* Xk  = (bf16_t*)(ws + SZX);
  bf16_t* Xv  = (bf16_t*)(ws + 2 * SZX);
  bf16_t* WT  = (bf16_t*)(ws + 3 * SZX);      // WqT,WkT,WvT,WoT consecutive
  bf16_t* WqT = WT;
  bf16_t* WkT = WT + NW;
  bf16_t* WvT = WT + 2 * NW;
  bf16_t* WoT = WT + 3 * NW;
  bf16_t* Qb  = (bf16_t*)(ws + 3 * SZX + 4 * SZW);
  bf16_t* Kb  = Qb + NX;
  bf16_t* Vb  = Kb + NX;
  bf16_t* Vt  = Xq;                            // reuse (inputs dead after projections)
  bf16_t* Ctx = Xk;                            // reuse

  float* out  = (float*)d_out;                       // [B,S,DM]
  float* attn = out + (size_t)B_ * S_ * DM;          // [B,H,S,S]

  const int THREADS = 256;

  // 1) weights -> bf16 transposed
  cvt_weight_T<<<(4 * (int)NW + THREADS - 1) / THREADS, THREADS, 0, stream>>>(Wq, Wk, Wv, Wo, WT);
  // 2) activations -> bf16
  cvt_f32_to_bf16<<<((int)NX + THREADS - 1) / THREADS, THREADS, 0, stream>>>(q, Xq, (int)NX);
  cvt_f32_to_bf16<<<((int)NX + THREADS - 1) / THREADS, THREADS, 0, stream>>>(k, Xk, (int)NX);
  cvt_f32_to_bf16<<<((int)NX + THREADS - 1) / THREADS, THREADS, 0, stream>>>(v, Xv, (int)NX);

  // 3) projections: [4096,768] x [768,768] -> bf16 (16x64 strip per wave)
  const int M = B_ * S_, N = DM, K = DM;
  const int gtiles  = (M / 16) * (N / (16 * GEMM_NT));  // 256 * 12 = 3072
  const int gblocks = gtiles / (THREADS / 32);          // 384
  gemm_bf16_kernel<<<gblocks, THREADS, 0, stream>>>(Xq, WqT, Qb, nullptr, M, N, K, 0);
  gemm_bf16_kernel<<<gblocks, THREADS, 0, stream>>>(Xk, WkT, Kb, nullptr, M, N, K, 0);
  gemm_bf16_kernel<<<gblocks, THREADS, 0, stream>>>(Xv, WvT, Vb, nullptr, M, N, K, 0);

  // 4) V -> Vt [B,H,DK,S]
  transpose_v_bf16<<<((int)NX + THREADS - 1) / THREADS, THREADS, 0, stream>>>(Vb, Vt);

  // 5) logits -> attn (fp32, masked & scaled)
  const int ltiles = B_ * NH * (S_ / 16) * (S_ / 16); // 262144
  logits_kernel<<<ltiles / (THREADS / 32), THREADS, 0, stream>>>(Qb, Kb, mask, attn);

  // 6) softmax in place, one block per row
  softmax_kernel<<<B_ * NH * S_, THREADS, 0, stream>>>(attn);

  // 7) AV -> Ctx bf16 (concat layout), 16x96 strip per wave
  const int atiles = B_ * NH * (S_ / 16);             // 2048
  av_kernel<<<atiles / (THREADS / 32), THREADS, 0, stream>>>(attn, Vt, Ctx);

  // 8) out = Ctx @ Wo (fp32 out)
  gemm_bf16_kernel<<<gblocks, THREADS, 0, stream>>>(Ctx, WoT, nullptr, out, M, N, K, 1);
}
